// DualOutputMoE_67242007986600
// MI455X (gfx1250) — compile-verified, matrix operand order checked by reading, not run on hardware
//
#include <hip/hip_runtime.h>
#include <hip/hip_bf16.h>
#include <math.h>

typedef __attribute__((ext_vector_type(16))) __bf16 v16bf;
typedef __attribute__((ext_vector_type(8)))  float  v8f;

#define T_TOK 4096
#define H_DIM 1024
#define E_NUM 16
#define F_DIM 4096

// workspace layout (bytes)
#define O_TOPI    0u          // int[T*2]
#define O_TOPW    32768u      // float[T*2]
#define O_COUNTS  65536u      // int[16]
#define O_CSUM    66560u      // float[16]
#define O_TOKENS  131072u     // int[16][4096]
#define O_WEIGHTS 393216u     // float[16][4096]
#define O_SVEC    655360u     // float[16][4096]
#define O_ACCH    917504u     // float[1024]
// fast-path packed buffers
#define O_XPHI    1048576u                       // bf16[T*H] permuted-K
#define O_XPLO    (O_XPHI + 8388608u)            // bf16[T*H]
#define O_W1HI    (O_XPLO + 8388608u)            // bf16[E*H*F] fragment-packed
#define O_W1LO    (O_W1HI + 134217728u)          // bf16[E*H*F]
#define O_END     (O_W1LO + 134217728u)          // ~274 MB

__device__ __forceinline__ int kperm(int i) {           // A/B fragment K permutation
    return (i < 8) ? i : (i < 16) ? i + 8 : (i < 24) ? i - 8 : i;
}

// ---------------------------------------------------------------- init
__global__ void moe_init(float* svec, float* accH) {
    int i = blockIdx.x * blockDim.x + threadIdx.x;
    int n = E_NUM * F_DIM + H_DIM;
    for (; i < n; i += gridDim.x * blockDim.x) {
        if (i < E_NUM * F_DIM) svec[i] = 0.0f;
        else                   accH[i - E_NUM * F_DIM] = 0.0f;
    }
}

// ---------------------------------------------------------------- routing
__global__ void moe_route(const float* __restrict__ x, const float* __restrict__ Wg,
                          const float* __restrict__ bg, int* __restrict__ topi,
                          float* __restrict__ topw) {
    int t = blockIdx.x * blockDim.x + threadIdx.x;
    if (t >= T_TOK) return;
    float sc[E_NUM];
#pragma unroll
    for (int e = 0; e < E_NUM; ++e) sc[e] = bg[e];
    const float* xr = x + (size_t)t * H_DIM;
    for (int h = 0; h < H_DIM; ++h) {
        float xv = xr[h];
        const float* wr = Wg + (size_t)h * E_NUM;
#pragma unroll
        for (int e = 0; e < E_NUM; ++e) sc[e] = fmaf(xv, wr[e], sc[e]);
    }
    float v1 = -1e30f, v2 = -1e30f; int i1 = 0, i2 = 0;
#pragma unroll
    for (int e = 0; e < E_NUM; ++e) {
        float v = sc[e];
        if (v > v1) { v2 = v1; i2 = i1; v1 = v; i1 = e; }
        else if (v > v2) { v2 = v; i2 = e; }
    }
    float e2 = __expf(v2 - v1);
    float inv = 1.0f / (1.0f + e2);
    topi[t * 2 + 0] = i1;  topi[t * 2 + 1] = i2;
    topw[t * 2 + 0] = inv; topw[t * 2 + 1] = e2 * inv;
}

// ---------------------------------------------------------------- compaction (deterministic)
__global__ void moe_compact(const int* __restrict__ topi, const float* __restrict__ topw,
                            int* __restrict__ tokens, float* __restrict__ weights,
                            int* __restrict__ counts, float* __restrict__ csum) {
    int e = blockIdx.x;
    int tid = threadIdx.x;
    __shared__ int   cnt[256];
    __shared__ float wsum[256];
    int base = tid * (T_TOK / 256);
    int c = 0; float ws = 0.0f;
    for (int i = 0; i < T_TOK / 256; ++i) {
        int t = base + i;
        if (topi[t * 2] == e)          { c++; ws += topw[t * 2]; }
        else if (topi[t * 2 + 1] == e) { c++; ws += topw[t * 2 + 1]; }
    }
    cnt[tid] = c; wsum[tid] = ws;
    __syncthreads();
    if (tid == 0) {
        int run = 0; float wt = 0.0f;
        for (int i = 0; i < 256; ++i) { int v = cnt[i]; cnt[i] = run; run += v; wt += wsum[i]; }
        counts[e] = run; csum[e] = wt;
    }
    __syncthreads();
    int off = cnt[tid];
    int* te = tokens + (size_t)e * T_TOK;
    float* we = weights + (size_t)e * T_TOK;
    for (int i = 0; i < T_TOK / 256; ++i) {
        int t = base + i;
        float w; bool hit = false;
        if (topi[t * 2] == e)          { w = topw[t * 2];     hit = true; }
        else if (topi[t * 2 + 1] == e) { w = topw[t * 2 + 1]; hit = true; }
        if (hit) { te[off] = t; we[off] = w; off++; }
    }
}

// ---------------------------------------------------------------- pack x -> bf16 hi/lo, K-permuted rows
__global__ void pack_x(const float* __restrict__ x, __bf16* __restrict__ xphi,
                       __bf16* __restrict__ xplo) {
    int i = blockIdx.x * blockDim.x + threadIdx.x;
    const int n = T_TOK * H_DIM;
    for (; i < n; i += gridDim.x * blockDim.x) {
        int w = i & 31;
        float v = x[(i & ~31) + kperm(w)];
        __bf16 h = (__bf16)v;
        xphi[i] = h;
        xplo[i] = (__bf16)(v - (float)h);
    }
}

// ---------------------------------------------------------------- pack W1 -> per-wave B fragments, bf16 hi/lo
// fragment layout: [(e*32+kt)*256 + ftile][lane][j], 512 bf16 per 32x16 fragment
__global__ void pack_w1(const float* __restrict__ W1, __bf16* __restrict__ w1hi,
                        __bf16* __restrict__ w1lo) {
    const int e = blockIdx.z, kt = blockIdx.y, fb = blockIdx.x;   // fb: 0..31 (128 cols)
    const int tid = threadIdx.x;
    __shared__ float sW[32 * 128];
    const float* src = W1 + (size_t)e * H_DIM * F_DIM + (size_t)(kt * 32) * F_DIM + fb * 128;
    for (int i = tid; i < 32 * 128; i += 256) {
        int k = i >> 7, f = i & 127;
        sW[i] = src[(size_t)k * F_DIM + f];
    }
    __syncthreads();
    const int frag = tid >> 5, lane = tid & 31;
    const int kb = (lane >> 4) * 8, fl = frag * 16 + (lane & 15);
    v16bf hv, lv;
#pragma unroll
    for (int j = 0; j < 16; ++j) {
        int kk = kb + ((j < 8) ? j : (j + 8));
        float v = sW[kk * 128 + fl];
        __bf16 h = (__bf16)v;
        hv[j] = h; lv[j] = (__bf16)(v - (float)h);
    }
    size_t fragidx = ((size_t)(e * 32 + kt) * 256 + (fb * 8 + frag));
    *(v16bf*)(w1hi + fragidx * 512 + lane * 16) = hv;
    *(v16bf*)(w1lo + fragidx * 512 + lane * 16) = lv;
}

// ---------------------------------------------------------------- FAST GEMM1: pure load+WMMA inner loop
__global__ void moe_ffn1_fast(const __bf16* __restrict__ xphi, const __bf16* __restrict__ xplo,
                              const __bf16* __restrict__ w1hi, const __bf16* __restrict__ w1lo,
                              const float* __restrict__ b1, const int* __restrict__ tokens,
                              const float* __restrict__ weights, const int* __restrict__ counts,
                              float* __restrict__ svec) {
    const int e = blockIdx.y;
    const int tid = threadIdx.x, lane = tid & 31, wave = tid >> 5;
    __shared__ int   sT[16];
    __shared__ float sC[16];
    const int n = counts[e];
    const int ntiles = (n + 15) >> 4;
    const int* te = tokens + (size_t)e * T_TOK;
    const float* we = weights + (size_t)e * T_TOK;
    const int m = lane & 15;
    const int ftile = blockIdx.x * 8 + wave;
    const int coln = ftile * 16 + m;
    const float bias = b1[(size_t)e * F_DIM + coln];
    const __bf16* bbh = w1hi + ((size_t)(e * 32) * 256 + ftile) * 512 + lane * 16;
    const __bf16* bbl = w1lo + ((size_t)(e * 32) * 256 + ftile) * 512 + lane * 16;
    const int aoff = (lane >> 4) * 16;

    float fsum = 0.0f;
    for (int mt = 0; mt < ntiles; ++mt) {
        __syncthreads();
        if (tid < 16) {
            int i = mt * 16 + tid;
            sT[tid] = (i < n) ? te[i] : te[0];
            sC[tid] = (i < n) ? we[i] : 0.0f;
        }
        __syncthreads();
        const __bf16* arh = xphi + (size_t)sT[m] * H_DIM + aoff;
        const __bf16* arl = xplo + (size_t)sT[m] * H_DIM + aoff;

        v8f acc = {};
#pragma unroll 4
        for (int kt = 0; kt < H_DIM / 32; ++kt) {
            v16bf ahi = *(const v16bf*)(arh + kt * 32);
            v16bf alo = *(const v16bf*)(arl + kt * 32);
            v16bf bhi = *(const v16bf*)(bbh + (size_t)kt * (256 * 512));
            v16bf blo = *(const v16bf*)(bbl + (size_t)kt * (256 * 512));
            acc = __builtin_amdgcn_wmma_f32_16x16x32_bf16(false, ahi, false, bhi, (short)0, acc, false, false);
            acc = __builtin_amdgcn_wmma_f32_16x16x32_bf16(false, ahi, false, blo, (short)0, acc, false, false);
            acc = __builtin_amdgcn_wmma_f32_16x16x32_bf16(false, alo, false, bhi, (short)0, acc, false, false);
        }

        float colsum = 0.0f;
#pragma unroll
        for (int r = 0; r < 8; ++r) {
            int mrow = r + 8 * (lane >> 4);
            float v = acc[r] + bias;
            float g = 0.5f * v * (1.0f + erff(v * 0.70710678118654752f));
            colsum += sC[mrow] * g;
        }
        colsum += __shfl_xor(colsum, 16);
        fsum += colsum;
    }
    if (lane < 16 && ntiles > 0)
        atomicAdd(&svec[(size_t)e * F_DIM + coln], fsum);
}

// ---------------------------------------------------------------- FALLBACK GEMM1 (in-loop conversion)
__global__ void moe_ffn1_slow(const float* __restrict__ x, const float* __restrict__ W1,
                              const float* __restrict__ b1, const int* __restrict__ tokens,
                              const float* __restrict__ weights, const int* __restrict__ counts,
                              float* __restrict__ svec) {
    const int e = blockIdx.y;
    const int f0 = blockIdx.x * 128;
    const int tid = threadIdx.x, lane = tid & 31, wave = tid >> 5;
    __shared__ float sA[16 * 32];
    __shared__ int   sT[16];
    __shared__ float sC[16];
    const int n = counts[e];
    const int ntiles = (n + 15) >> 4;
    const float* w1e = W1 + (size_t)e * H_DIM * F_DIM;
    const int* te = tokens + (size_t)e * T_TOK;
    const float* we = weights + (size_t)e * T_TOK;
    const int m = lane & 15;
    const int kb = (lane >> 4) * 8;
    const int coln = f0 + wave * 16 + m;
    const float bias = b1[(size_t)e * F_DIM + coln];
    float fsum = 0.0f;

    for (int mt = 0; mt < ntiles; ++mt) {
        __syncthreads();
        if (tid < 16) {
            int i = mt * 16 + tid;
            sT[tid] = (i < n) ? te[i] : te[0];
            sC[tid] = (i < n) ? we[i] : 0.0f;
        }
        __syncthreads();
        v8f acc = {};
        for (int kt = 0; kt < H_DIM / 32; ++kt) {
            const int k0 = kt * 32;
            __syncthreads();
            {
                int idx = tid * 2;
                int row = idx >> 5, col = idx & 31;
                const float* xr = x + (size_t)sT[row] * H_DIM + k0 + col;
                sA[idx] = xr[0]; sA[idx + 1] = xr[1];
            }
            __syncthreads();
            v16bf ahi, alo, bhi, blo;
#pragma unroll
            for (int j = 0; j < 16; ++j) {
                const int kk = kb + ((j < 8) ? j : (j + 8));
                float av = sA[m * 32 + kk];
                __bf16 ah = (__bf16)av;
                ahi[j] = ah; alo[j] = (__bf16)(av - (float)ah);
                float bv = w1e[(size_t)(k0 + kk) * F_DIM + coln];
                __bf16 bh = (__bf16)bv;
                bhi[j] = bh; blo[j] = (__bf16)(bv - (float)bh);
            }
            acc = __builtin_amdgcn_wmma_f32_16x16x32_bf16(false, ahi, false, bhi, (short)0, acc, false, false);
            acc = __builtin_amdgcn_wmma_f32_16x16x32_bf16(false, ahi, false, blo, (short)0, acc, false, false);
            acc = __builtin_amdgcn_wmma_f32_16x16x32_bf16(false, alo, false, bhi, (short)0, acc, false, false);
        }
        float colsum = 0.0f;
#pragma unroll
        for (int r = 0; r < 8; ++r) {
            int mrow = r + 8 * (lane >> 4);
            float v = acc[r] + bias;
            float g = 0.5f * v * (1.0f + erff(v * 0.70710678118654752f));
            colsum += sC[mrow] * g;
        }
        colsum += __shfl_xor(colsum, 16);
        fsum += colsum;
    }
    if (lane < 16 && ntiles > 0)
        atomicAdd(&svec[(size_t)e * F_DIM + coln], fsum);
}

// ---------------------------------------------------------------- GEMV over W2
__global__ void moe_gemv2(const float* __restrict__ svec, const float* __restrict__ W2,
                          float* __restrict__ accH) {
    const int e  = blockIdx.z;
    const int h  = blockIdx.y * 256 + threadIdx.x;
    const int f0 = blockIdx.x * 256;
    const float* w2e = W2 + (size_t)e * F_DIM * H_DIM;
    const float* sv  = svec + (size_t)e * F_DIM;
    float a = 0.0f;
#pragma unroll 4
    for (int f = f0; f < f0 + 256; ++f)
        a = fmaf(sv[f], w2e[(size_t)f * H_DIM + h], a);
    atomicAdd(&accH[h], a);
}

// ---------------------------------------------------------------- finalize
__global__ void moe_final(const float* __restrict__ accH, const float* __restrict__ csum,
                          const float* __restrict__ b2, float* __restrict__ out) {
    int h = blockIdx.x * blockDim.x + threadIdx.x;
    if (h >= H_DIM) return;
    float v = accH[h];
    float tot = 0.0f;
#pragma unroll
    for (int e = 0; e < E_NUM; ++e) {
        tot += csum[e];
        v = fmaf(csum[e], b2[(size_t)e * H_DIM + h], v);
    }
    out[h] = v / tot;
}

// ---------------------------------------------------------------- launch
extern "C" void kernel_launch(void* const* d_in, const int* in_sizes, int n_in,
                              void* d_out, int out_size, void* d_ws, size_t ws_size,
                              hipStream_t stream) {
    const float* x  = (const float*)d_in[0];
    const float* Wg = (const float*)d_in[1];
    const float* bg = (const float*)d_in[2];
    const float* W1 = (const float*)d_in[3];
    const float* b1 = (const float*)d_in[4];
    const float* W2 = (const float*)d_in[5];
    const float* b2 = (const float*)d_in[6];
    float* out = (float*)d_out;

    char* ws = (char*)d_ws;
    int*    topi    = (int*)   (ws + O_TOPI);
    float*  topw    = (float*) (ws + O_TOPW);
    int*    counts  = (int*)   (ws + O_COUNTS);
    float*  csum    = (float*) (ws + O_CSUM);
    int*    tokens  = (int*)   (ws + O_TOKENS);
    float*  weights = (float*) (ws + O_WEIGHTS);
    float*  svec    = (float*) (ws + O_SVEC);
    float*  accH    = (float*) (ws + O_ACCH);
    __bf16* xphi    = (__bf16*)(ws + O_XPHI);
    __bf16* xplo    = (__bf16*)(ws + O_XPLO);
    __bf16* w1hi    = (__bf16*)(ws + O_W1HI);
    __bf16* w1lo    = (__bf16*)(ws + O_W1LO);

    moe_init<<<dim3(64), dim3(256), 0, stream>>>(svec, accH);
    moe_route<<<dim3(T_TOK / 256), dim3(256), 0, stream>>>(x, Wg, bg, topi, topw);
    moe_compact<<<dim3(E_NUM), dim3(256), 0, stream>>>(topi, topw, tokens, weights, counts, csum);

    if (ws_size >= (size_t)O_END) {
        pack_x<<<dim3(1024), dim3(256), 0, stream>>>(x, xphi, xplo);
        pack_w1<<<dim3(32, 32, 16), dim3(256), 0, stream>>>(W1, w1hi, w1lo);
        moe_ffn1_fast<<<dim3(F_DIM / 128, E_NUM), dim3(256), 0, stream>>>(
            xphi, xplo, w1hi, w1lo, b1, tokens, weights, counts, svec);
    } else {
        moe_ffn1_slow<<<dim3(F_DIM / 128, E_NUM), dim3(256), 0, stream>>>(
            x, W1, b1, tokens, weights, counts, svec);
    }

    moe_gemv2<<<dim3(F_DIM / 256, H_DIM / 256, E_NUM), dim3(256), 0, stream>>>(svec, W2, accH);
    moe_final<<<dim3(H_DIM / 256), dim3(256), 0, stream>>>(accH, csum, b2, out);

    (void)in_sizes; (void)n_in; (void)out_size; (void)ws_size;
}
